// MPB_19782619365720
// MI455X (gfx1250) — matrix operand order repaired and checked by reading, not compiled, not run on hardware
//
#include <hip/hip_runtime.h>
#include <hip/hip_bf16.h>
#include <cstdint>
#include <cstddef>

#define DEV __device__ __forceinline__

typedef __attribute__((ext_vector_type(16))) __bf16 bf16x16;
typedef __attribute__((ext_vector_type(8)))  float  v8f;

union AFrag { uint4 u[2]; bf16x16 v; };

static constexpr int NP = 16 * 64 * 64;      // 65536 pixels (B*H*W)
static constexpr float SELU_SCALE = 1.0507009873554805f;
static constexpr float SELU_ALPHA = 1.6732632423543772f;

DEV float bf2f(unsigned short h) {
  union { unsigned int u; float f; } x; x.u = ((unsigned int)h) << 16; return x.f;
}
DEV unsigned short f2bf(float f) {
  union { float f; unsigned int u; } x; x.f = f;
  unsigned int r = x.u + 0x7fffu + ((x.u >> 16) & 1u);
  return (unsigned short)(r >> 16);
}
DEV float seluf(float x) { return SELU_SCALE * (x > 0.f ? x : SELU_ALPHA * (__expf(x) - 1.f)); }
DEV int iclampi(int x, int lo, int hi) { return x < lo ? lo : (x > hi ? hi : x); }

// ---------------- pack kernels ----------------

// f32 NCHW (C=64) -> bf16 NHWC
__global__ __launch_bounds__(256) void k_pack_in(const float* __restrict__ src,
                                                 unsigned short* __restrict__ dst) {
  int t = blockIdx.x * 256 + threadIdx.x;           // t = pix*64 + c
  if (t >= NP * 64) return;
  int pix = t >> 6, c = t & 63;
  int n = pix >> 12, hw = pix & 4095;
  dst[t] = f2bf(src[(((size_t)n * 64 + c) << 12) + hw]);
}

// weights f32 (O,I,ks,ks) -> bf16 [COUTP][KTOTP], K order = tap*I + ci, zero padded
__global__ __launch_bounds__(256) void k_pack_w(const float* __restrict__ w,
                                                unsigned short* __restrict__ Wt,
                                                int O, int I, int KS_, int KTOTP, int COUTP) {
  int t = blockIdx.x * 256 + threadIdx.x;
  if (t >= COUTP * KTOTP) return;
  int col = t / KTOTP, k = t - col * KTOTP;
  int tap = k / I, ci = k - tap * I;
  float v = 0.f;
  if (col < O && tap < KS_ * KS_)
    v = w[((size_t)(col * I + ci) * KS_ + (tap / KS_)) * KS_ + (tap % KS_)];
  Wt[t] = f2bf(v);
}

// db4 weights (1600,64) reordered tap-major: col q' = tap*64 + c  <-  o = c*25 + tap
__global__ __launch_bounds__(256) void k_pack_db4(const float* __restrict__ w4,
                                                  const float* __restrict__ b4,
                                                  unsigned short* __restrict__ Wt4,
                                                  float* __restrict__ b4r) {
  int t = blockIdx.x * 256 + threadIdx.x;
  if (t < 1600) { int tap = t >> 6, c = t & 63; b4r[t] = b4[c * 25 + tap]; }
  if (t >= 1600 * 64) return;
  int q = t >> 6, d = t & 63;
  int tap = q >> 6, c = q & 63;
  Wt4[t] = f2bf(w4[(size_t)(c * 25 + tap) * 64 + d]);
}

// ---------------- generic conv-as-GEMM with WMMA ----------------
// A fragment group load: 8 contiguous bf16 channels of one tap pixel (zero pad OOB)
template<int CIN, int KS>
DEV uint4 loadA(const unsigned short* __restrict__ A, int a_stride, int a_coff,
                int n, int h, int w, int k) {
  uint4 z; z.x = 0u; z.y = 0u; z.z = 0u; z.w = 0u;
  const int tap = k / CIN;
  const int c = k - tap * CIN;
  if (tap >= KS * KS) return z;                        // K padding
  const int di = tap / KS - KS / 2;
  const int dj = tap % KS - KS / 2;
  const int h2 = h + di, w2 = w + dj;
  if ((unsigned)h2 >= 64u || (unsigned)w2 >= 64u) return z;  // SAME zero pad
  const int pix = (n << 12) + (h2 << 6) + w2;
  return *(const uint4*)(A + (size_t)pix * a_stride + a_coff + c);
}

// EPI: 0 = store bf16 NHWC
//      1 = mlp: scatter into dmap (SEL=0 img -> also store img_buf; SEL=1 degra)
//      2 = denoise: v+img, d_out += w1*v
//      3 = dehaze:  v*(img-a)+a, d_out += w2*...
//      4 = dedark:  img*mul+v,   d_out += w3*...
template<int CIN, int COUT, int KS, bool ACT_SELU, int EPI, int SEL>
__global__ __launch_bounds__(256) void k_convgemm(
    const unsigned short* __restrict__ A, int a_stride, int a_coff,
    const unsigned short* __restrict__ Wt, const float* __restrict__ bias,
    unsigned short* __restrict__ dst, int d_stride,
    unsigned short* __restrict__ dmap,
    float* __restrict__ out,
    const float* __restrict__ wsel,
    const unsigned short* __restrict__ imgb,
    const float* __restrict__ abuf,
    const unsigned short* __restrict__ mulb)
{
  constexpr int KTOT  = KS * KS * CIN;
  constexpr int KTOTP = (KTOT + 31) & ~31;
  constexpr int NT    = (COUT + 15) / 16;

  const int lane = threadIdx.x & 31;
  const int wv   = threadIdx.x >> 5;
  const int pixbase = (blockIdx.x * 8 + wv) * 16;
  const int row = lane & 15, hi = lane >> 4;
  const int p = pixbase + row;
  const int n = p >> 12, h = (p >> 6) & 63, w = p & 63;

  v8f acc[NT];
#pragma unroll
  for (int i = 0; i < NT; i++) acc[i] = (v8f)(0.f);

  for (int kb = 0; kb < KTOTP; kb += 32) {
    AFrag a;
    a.u[0] = loadA<CIN, KS>(A, a_stride, a_coff, n, h, w, kb + hi * 8);
    a.u[1] = loadA<CIN, KS>(A, a_stride, a_coff, n, h, w, kb + 16 + hi * 8);
#pragma unroll
    for (int nt = 0; nt < NT; nt++) {
      AFrag b;
      const uint4* bp = (const uint4*)(Wt + (size_t)(nt * 16 + row) * KTOTP + kb);
      b.u[0] = bp[hi]; b.u[1] = bp[2 + hi];
      acc[nt] = __builtin_amdgcn_wmma_f32_16x16x32_bf16(
          false, a.v, false, b.v, (short)0, acc[nt], false, false);
    }
  }

#pragma unroll
  for (int nt = 0; nt < NT; nt++) {
    const int c = nt * 16 + row;
    if (c >= COUT) continue;
    const float bv = bias[c];
#pragma unroll
    for (int r = 0; r < 8; r++) {
      const int m = r + hi * 8;
      const int pix = pixbase + m;
      float v = acc[nt][r] + bv;
      if (ACT_SELU) v = seluf(v);
      if (EPI == 0) {
        dst[(size_t)pix * d_stride + c] = f2bf(v);
      } else if (EPI == 1) {
        const int cp = (c < 32) ? (2 * c + SEL) : (64 + 2 * (c - 32) + SEL);
        dmap[(size_t)pix * 128 + cp] = f2bf(v);
        if (SEL == 0) dst[(size_t)pix * 64 + c] = f2bf(v);
      } else {
        const int nn = pix >> 12, hw = pix & 4095;
        const size_t oidx = (((size_t)nn * 64 + c) << 12) + hw;
        if (EPI == 2) {
          const float ov = v + bf2f(imgb[(size_t)pix * 64 + c]);
          out[oidx] += wsel[pix * 4 + 1] * ov;
        } else if (EPI == 3) {
          const float av = abuf[nn * 64 + c];
          const float iv = bf2f(imgb[(size_t)pix * 64 + c]);
          out[oidx] += wsel[pix * 4 + 2] * (v * (iv - av) + av);
        } else {
          const float iv = bf2f(imgb[(size_t)pix * 64 + c]);
          const float mv = bf2f(mulb[(size_t)pix * 64 + c]);
          out[oidx] += wsel[pix * 4 + 3] * (iv * mv + v);
        }
      }
    }
  }
}

// ---------------- fused db4 GEMM + 5x5 dynamic (edge-padded) conv, writes w0*deblur ----------------
__global__ __launch_bounds__(256) void k_deblur(
    const unsigned short* __restrict__ xdb,   // [NP][64] bf16
    const unsigned short* __restrict__ Wt4,   // [25*64][64] bf16, tap-major cols
    const float* __restrict__ b4r,            // [1600] reordered bias
    const unsigned short* __restrict__ imgb,  // [NP][64] bf16
    const float* __restrict__ wsel,
    float* __restrict__ out)
{
  const int lane = threadIdx.x & 31;
  const int wv   = threadIdx.x >> 5;
  const int task = blockIdx.x * 8 + wv;
  const int pt = task >> 2, ct = task & 3;
  const int pixbase = pt * 16, c0 = ct * 16;
  const int row = lane & 15, hi = lane >> 4;
  const int n = pixbase >> 12, h = (pixbase >> 6) & 63, w0 = pixbase & 63;

  // A: x rows (K=64), invariant across taps
  const uint4* ap = (const uint4*)(xdb + (size_t)(pixbase + row) * 64);
  AFrag a0, a1;
  a0.u[0] = ap[hi];     a0.u[1] = ap[2 + hi];
  a1.u[0] = ap[4 + hi]; a1.u[1] = ap[6 + hi];

  const int cl = c0 + row;
  float outacc[8];
#pragma unroll
  for (int r = 0; r < 8; r++) outacc[r] = 0.f;

  for (int tap = 0; tap < 25; tap++) {
    const int col = tap * 64 + cl;
    const uint4* bp = (const uint4*)(Wt4 + (size_t)col * 64);
    AFrag b0, b1;
    b0.u[0] = bp[hi];     b0.u[1] = bp[2 + hi];
    b1.u[0] = bp[4 + hi]; b1.u[1] = bp[6 + hi];
    v8f bk = (v8f)(0.f);
    bk = __builtin_amdgcn_wmma_f32_16x16x32_bf16(false, a0.v, false, b0.v, (short)0, bk, false, false);
    bk = __builtin_amdgcn_wmma_f32_16x16x32_bf16(false, a1.v, false, b1.v, (short)0, bk, false, false);
    const float bias = b4r[col];
    const int di = tap / 5 - 2, dj = tap % 5 - 2;
    const int h2 = iclampi(h + di, 0, 63);
#pragma unroll
    for (int r = 0; r < 8; r++) {
      const int m = r + hi * 8;
      const int w2 = iclampi(w0 + m + dj, 0, 63);
      const float iv = bf2f(imgb[((size_t)((n << 12) + (h2 << 6) + w2)) * 64 + cl]);
      outacc[r] += (bk[r] + bias) * iv;
    }
  }
#pragma unroll
  for (int r = 0; r < 8; r++) {
    const int pix = pixbase + r + hi * 8;
    out[(((size_t)n * 64 + cl) << 12) + (pix & 4095)] = wsel[pix * 4 + 0] * outacc[r];
  }
}

// ---------------- small scalar kernels ----------------

__global__ __launch_bounds__(256) void k_pred2(const unsigned short* __restrict__ p1,
                                               const float* __restrict__ w2,
                                               const float* __restrict__ b2,
                                               float* __restrict__ wsel) {
  __shared__ float sw[6 * 64];
  __shared__ float sb[8];
  const int t = threadIdx.x;
  for (int i = t; i < 384; i += 256) sw[i] = w2[i];
  if (t < 6) sb[t] = b2[t];
  __syncthreads();
  const int pix = blockIdx.x * 256 + t;
  const unsigned short* rowp = p1 + (size_t)pix * 64;
  float a6[6];
#pragma unroll
  for (int j = 0; j < 6; j++) a6[j] = sb[j];
  for (int c = 0; c < 64; c++) {
    const float x = bf2f(rowp[c]);
#pragma unroll
    for (int j = 0; j < 6; j++) a6[j] += x * sw[j * 64 + c];
  }
  float mx = a6[0];
#pragma unroll
  for (int j = 1; j < 6; j++) mx = fmaxf(mx, a6[j]);
  float s = 0.f;
#pragma unroll
  for (int j = 0; j < 6; j++) { a6[j] = __expf(a6[j] - mx); s += a6[j]; }
  const float inv = 1.f / s;
#pragma unroll
  for (int j = 0; j < 4; j++) wsel[pix * 4 + j] = a6[j] * inv;
}

// mean over H*W of dmap "second" slice -> meanbuf[n][c]
__global__ __launch_bounds__(256) void k_mean2(const unsigned short* __restrict__ dmap,
                                               float* __restrict__ meanbuf) {
  const int n = blockIdx.x >> 6, c = blockIdx.x & 63;
  __shared__ float red[256];
  float s = 0.f;
  for (int p = threadIdx.x; p < 4096; p += 256)
    s += bf2f(dmap[((size_t)((n << 12) + p)) * 128 + 64 + c]);
  red[threadIdx.x] = s;
  __syncthreads();
  for (int st = 128; st > 0; st >>= 1) {
    if (threadIdx.x < st) red[threadIdx.x] += red[threadIdx.x + st];
    __syncthreads();
  }
  if (threadIdx.x == 0) meanbuf[n * 64 + c] = red[0] * (1.f / 4096.f);
}

// dha on 1x1 spatial: only center taps of the 3x3 weights contribute (SAME zero pad)
__global__ void k_dha(const float* __restrict__ meanbuf,
                      const float* __restrict__ w1, const float* __restrict__ b1,
                      const float* __restrict__ w2, const float* __restrict__ b2,
                      float* __restrict__ abuf) {
  const int n = blockIdx.x, t = threadIdx.x;   // 64 threads
  __shared__ float m2[64], a0[8];
  m2[t] = meanbuf[n * 64 + t];
  __syncthreads();
  if (t < 8) {
    float s = b1[t];
    for (int j = 0; j < 64; j++) s += m2[j] * w1[(t * 64 + j) * 9 + 4];
    a0[t] = seluf(s);
  }
  __syncthreads();
  float s = b2[t];
  for (int i = 0; i < 8; i++) s += a0[i] * w2[(t * 8 + i) * 9 + 4];
  abuf[n * 64 + t] = seluf(s);
}

// ---------------- host launcher ----------------

extern "C" void kernel_launch(void* const* d_in, const int* in_sizes, int n_in,
                              void* d_out, int out_size, void* d_ws, size_t ws_size,
                              hipStream_t stream) {
  (void)in_sizes; (void)n_in; (void)out_size; (void)ws_size;
  const float* in_img    = (const float*)d_in[0];
  const float* in_degra  = (const float*)d_in[1];
  const float* in_prompt = (const float*)d_in[2];
  // params pytree: layers alphabetical, (b, w) per layer
  enum { L_db1 = 0, L_db2, L_db3, L_db4, L_dha1, L_dha2, L_dht1, L_dht2, L_dht3,
         L_dka1, L_dka2, L_dka3, L_dkm1, L_dkm2, L_dkm3, L_dn1, L_dn2, L_dn3,
         L_mlp_degra, L_mlp_img, L_pred1, L_pred2 };
  auto PB = [&](int i) { return (const float*)d_in[3 + 2 * i]; };
  auto PW = [&](int i) { return (const float*)d_in[3 + 2 * i + 1]; };

  char* ws = (char*)d_ws;
  size_t off = 0;
  auto alloc = [&](size_t bytes) -> void* {
    off = (off + 255) & ~(size_t)255;
    void* p = ws + off;
    off += bytes;
    return p;
  };

  unsigned short* img_nhwc = (unsigned short*)alloc((size_t)NP * 64 * 2);
  unsigned short* deg_nhwc = (unsigned short*)alloc((size_t)NP * 64 * 2);
  unsigned short* prm_nhwc = (unsigned short*)alloc((size_t)NP * 64 * 2);
  unsigned short* dmap     = (unsigned short*)alloc((size_t)NP * 128 * 2);
  unsigned short* img_buf  = (unsigned short*)alloc((size_t)NP * 64 * 2);
  unsigned short* p1buf    = (unsigned short*)alloc((size_t)NP * 64 * 2);
  float*          wsel     = (float*)alloc((size_t)NP * 4 * 4);
  unsigned short* t1       = (unsigned short*)alloc((size_t)NP * 64 * 2);
  unsigned short* t2       = (unsigned short*)alloc((size_t)NP * 8 * 2);
  unsigned short* xdb      = (unsigned short*)alloc((size_t)NP * 64 * 2);
  unsigned short* mulb     = (unsigned short*)alloc((size_t)NP * 64 * 2);
  float*          meanbuf  = (float*)alloc(16 * 64 * 4);
  float*          abuf     = (float*)alloc(16 * 64 * 4);

  auto wb = [](int O, int I, int KS_) {
    int KTOTP = (KS_ * KS_ * I + 31) & ~31;
    int COUTP = (O + 15) & ~15;
    return (size_t)KTOTP * COUTP * 2;
  };
  unsigned short* Wmlpimg = (unsigned short*)alloc(wb(64, 64, 1));
  unsigned short* Wmlpdeg = (unsigned short*)alloc(wb(64, 64, 1));
  unsigned short* Wpred1  = (unsigned short*)alloc(wb(64, 64, 1));
  unsigned short* Wdb1    = (unsigned short*)alloc(wb(64, 128, 3));
  unsigned short* Wdb2    = (unsigned short*)alloc(wb(8, 64, 3));
  unsigned short* Wdb3    = (unsigned short*)alloc(wb(64, 8, 3));
  unsigned short* Wdn1    = (unsigned short*)alloc(wb(64, 64, 3));
  unsigned short* Wdn2    = (unsigned short*)alloc(wb(8, 64, 3));
  unsigned short* Wdn3    = (unsigned short*)alloc(wb(64, 8, 3));
  unsigned short* Wdht1   = (unsigned short*)alloc(wb(64, 64, 3));
  unsigned short* Wdht2   = (unsigned short*)alloc(wb(8, 64, 3));
  unsigned short* Wdht3   = (unsigned short*)alloc(wb(64, 8, 3));
  unsigned short* Wdkm1   = (unsigned short*)alloc(wb(64, 64, 3));
  unsigned short* Wdkm2   = (unsigned short*)alloc(wb(8, 64, 3));
  unsigned short* Wdkm3   = (unsigned short*)alloc(wb(64, 8, 3));
  unsigned short* Wdka1   = (unsigned short*)alloc(wb(64, 64, 3));
  unsigned short* Wdka2   = (unsigned short*)alloc(wb(8, 64, 3));
  unsigned short* Wdka3   = (unsigned short*)alloc(wb(64, 8, 3));
  unsigned short* Wdb4    = (unsigned short*)alloc((size_t)1600 * 64 * 2);
  float*          b4r     = (float*)alloc(1600 * 4);

  float* out = (float*)d_out;

  // --- pack inputs ---
  {
    const int g = (NP * 64 + 255) / 256;
    k_pack_in<<<g, 256, 0, stream>>>(in_img, img_nhwc);
    k_pack_in<<<g, 256, 0, stream>>>(in_degra, deg_nhwc);
    k_pack_in<<<g, 256, 0, stream>>>(in_prompt, prm_nhwc);
  }
  // --- pack weights ---
  auto packw = [&](const float* w, unsigned short* Wt, int O, int I, int KS_) {
    int KTOTP = (KS_ * KS_ * I + 31) & ~31;
    int COUTP = (O + 15) & ~15;
    int elems = COUTP * KTOTP;
    k_pack_w<<<(elems + 255) / 256, 256, 0, stream>>>(w, Wt, O, I, KS_, KTOTP, COUTP);
  };
  packw(PW(L_mlp_img),   Wmlpimg, 64, 64, 1);
  packw(PW(L_mlp_degra), Wmlpdeg, 64, 64, 1);
  packw(PW(L_pred1),     Wpred1,  64, 64, 1);
  packw(PW(L_db1),  Wdb1,  64, 128, 3);
  packw(PW(L_db2),  Wdb2,   8,  64, 3);
  packw(PW(L_db3),  Wdb3,  64,   8, 3);
  packw(PW(L_dn1),  Wdn1,  64,  64, 3);
  packw(PW(L_dn2),  Wdn2,   8,  64, 3);
  packw(PW(L_dn3),  Wdn3,  64,   8, 3);
  packw(PW(L_dht1), Wdht1, 64,  64, 3);
  packw(PW(L_dht2), Wdht2,  8,  64, 3);
  packw(PW(L_dht3), Wdht3, 64,   8, 3);
  packw(PW(L_dkm1), Wdkm1, 64,  64, 3);
  packw(PW(L_dkm2), Wdkm2,  8,  64, 3);
  packw(PW(L_dkm3), Wdkm3, 64,   8, 3);
  packw(PW(L_dka1), Wdka1, 64,  64, 3);
  packw(PW(L_dka2), Wdka2,  8,  64, 3);
  packw(PW(L_dka3), Wdka3, 64,   8, 3);
  k_pack_db4<<<(1600 * 64 + 255) / 256, 256, 0, stream>>>(PW(L_db4), PB(L_db4), Wdb4, b4r);

  const int GG = NP / 128;  // 512 blocks, 8 waves * 16-pixel tiles
  unsigned short* NU16 = nullptr; float* NUF = nullptr;

  // --- mlps: scatter into dmap (img also to img_buf) ---
  k_convgemm<64, 64, 1, false, 1, 0><<<GG, 256, 0, stream>>>(
      img_nhwc, 64, 0, Wmlpimg, PB(L_mlp_img), img_buf, 64, dmap, NUF, NUF, NU16, NUF, NU16);
  k_convgemm<64, 64, 1, false, 1, 1><<<GG, 256, 0, stream>>>(
      deg_nhwc, 64, 0, Wmlpdeg, PB(L_mlp_degra), NU16, 64, dmap, NUF, NUF, NU16, NUF, NU16);

  // --- prediction weights ---
  k_convgemm<64, 64, 1, false, 0, 0><<<GG, 256, 0, stream>>>(
      prm_nhwc, 64, 0, Wpred1, PB(L_pred1), p1buf, 64, NU16, NUF, NUF, NU16, NUF, NU16);
  k_pred2<<<NP / 256, 256, 0, stream>>>(p1buf, PW(L_pred2), PB(L_pred2), wsel);

  // --- deblur branch: db1 -> db2 -> db3 -> fused db4 + dynamic 5x5 (initializes d_out) ---
  k_convgemm<128, 64, 3, false, 0, 0><<<GG, 256, 0, stream>>>(
      dmap, 128, 0, Wdb1, PB(L_db1), t1, 64, NU16, NUF, NUF, NU16, NUF, NU16);
  k_convgemm<64, 8, 3, true, 0, 0><<<GG, 256, 0, stream>>>(
      t1, 64, 0, Wdb2, PB(L_db2), t2, 8, NU16, NUF, NUF, NU16, NUF, NU16);
  k_convgemm<8, 64, 3, true, 0, 0><<<GG, 256, 0, stream>>>(
      t2, 8, 0, Wdb3, PB(L_db3), xdb, 64, NU16, NUF, NUF, NU16, NUF, NU16);
  k_deblur<<<NP * 4 / 16 / 8, 256, 0, stream>>>(xdb, Wdb4, b4r, img_buf, wsel, out);

  // --- dha (needed by dehaze epilogue) ---
  k_mean2<<<16 * 64, 256, 0, stream>>>(dmap, meanbuf);
  k_dha<<<16, 64, 0, stream>>>(meanbuf, PW(L_dha1), PB(L_dha1), PW(L_dha2), PB(L_dha2), abuf);

  // --- denoise branch (input = second): out += w1 * (route3 + img) ---
  k_convgemm<64, 64, 3, false, 0, 0><<<GG, 256, 0, stream>>>(
      dmap, 128, 64, Wdn1, PB(L_dn1), t1, 64, NU16, NUF, NUF, NU16, NUF, NU16);
  k_convgemm<64, 8, 3, true, 0, 0><<<GG, 256, 0, stream>>>(
      t1, 64, 0, Wdn2, PB(L_dn2), t2, 8, NU16, NUF, NUF, NU16, NUF, NU16);
  k_convgemm<8, 64, 3, true, 2, 0><<<GG, 256, 0, stream>>>(
      t2, 8, 0, Wdn3, PB(L_dn3), NU16, 64, NU16, out, wsel, img_buf, NUF, NU16);

  // --- dehaze branch (input = first): out += w2 * (t*(img-a)+a) ---
  k_convgemm<64, 64, 3, false, 0, 0><<<GG, 256, 0, stream>>>(
      dmap, 128, 0, Wdht1, PB(L_dht1), t1, 64, NU16, NUF, NUF, NU16, NUF, NU16);
  k_convgemm<64, 8, 3, true, 0, 0><<<GG, 256, 0, stream>>>(
      t1, 64, 0, Wdht2, PB(L_dht2), t2, 8, NU16, NUF, NUF, NU16, NUF, NU16);
  k_convgemm<8, 64, 3, true, 3, 0><<<GG, 256, 0, stream>>>(
      t2, 8, 0, Wdht3, PB(L_dht3), NU16, 64, NU16, out, wsel, img_buf, abuf, NU16);

  // --- dedark: mul = route3(first, dkm) ; out += w3 * (img*mul + route3(second, dka)) ---
  k_convgemm<64, 64, 3, false, 0, 0><<<GG, 256, 0, stream>>>(
      dmap, 128, 0, Wdkm1, PB(L_dkm1), t1, 64, NU16, NUF, NUF, NU16, NUF, NU16);
  k_convgemm<64, 8, 3, true, 0, 0><<<GG, 256, 0, stream>>>(
      t1, 64, 0, Wdkm2, PB(L_dkm2), t2, 8, NU16, NUF, NUF, NU16, NUF, NU16);
  k_convgemm<8, 64, 3, true, 0, 0><<<GG, 256, 0, stream>>>(
      t2, 8, 0, Wdkm3, PB(L_dkm3), mulb, 64, NU16, NUF, NUF, NU16, NUF, NU16);
  k_convgemm<64, 64, 3, false, 0, 0><<<GG, 256, 0, stream>>>(
      dmap, 128, 64, Wdka1, PB(L_dka1), t1, 64, NU16, NUF, NUF, NU16, NUF, NU16);
  k_convgemm<64, 8, 3, true, 0, 0><<<GG, 256, 0, stream>>>(
      t1, 64, 0, Wdka2, PB(L_dka2), t2, 8, NU16, NUF, NUF, NU16, NUF, NU16);
  k_convgemm<8, 64, 3, true, 4, 0><<<GG, 256, 0, stream>>>(
      t2, 8, 0, Wdka3, PB(L_dka3), NU16, 64, NU16, out, wsel, img_buf, NUF, mulb);
}
